// DenseConv2d_full_68118181314793
// MI455X (gfx1250) — compile-verified, matrix operand order checked by reading, not run on hardware
//
#include <hip/hip_runtime.h>
#include <hip/hip_bf16.h>

// Problem constants (from setup_inputs)
#define BB 8
#define CR 8
#define KH 16
#define KW 8
#define CI 16
#define HH 256
#define WW 512
#define OC 64
// Merged kernel: 18 x 10 taps per (oc,ci), padded K per ci = 192 (6 chunks of 32)
#define KTAP 180
#define KPAD 192
#define VARSZ ((size_t)BB * CI * OC * KPAD)   // elements per K2 variant (bf16)

typedef __attribute__((ext_vector_type(16))) __bf16 v16bf;
typedef __attribute__((ext_vector_type(8)))  __bf16 v8bf;
typedef __attribute__((ext_vector_type(8)))  float  v8f;
typedef __attribute__((ext_vector_type(4)))  unsigned v4u;
typedef __attribute__((ext_vector_type(8)))  int v8i;
typedef __attribute__((ext_vector_type(4)))  int v4i;

// ---------------------------------------------------------------------------
// Phase A: build merged kernels K2 (4 variants: full, drop-u+, drop-v+, both)
//   var0 layout (for WMMA A-frags):  [b][ci][oc][KPAD]   (k contiguous)
//   var1..3 layout (edge kernel):    [b][ci][k][OC]      (oc contiguous)
// ---------------------------------------------------------------------------
__global__ __launch_bounds__(256) void k2_kernel(const float* __restrict__ resp,
                                                 const float* __restrict__ w,
                                                 __bf16* __restrict__ ws) {
  const int b  = blockIdx.x >> 4;
  const int ci = blockIdx.x & 15;
  for (int slot = threadIdx.x; slot < OC * KPAD; slot += 256) {
    const int oc = slot / KPAD;
    const int k  = slot % KPAD;
    float sF = 0.f, sU = 0.f, sV = 0.f, sUV = 0.f;
    if (k < KTAP) {
      const int i = k / 10;
      const int j = k % 10;
      for (int cr = 0; cr < CR; ++cr) {
        const float* wp = w + (size_t)(oc * (CI * CR) + ci * CR + cr) * 9;
        const float* rp = resp + (size_t)(b * CR + cr) * (KH * KW);
        #pragma unroll
        for (int up = 0; up < 3; ++up) {
          const int ri = i + up - 2;
          if (ri < 0 || ri >= KH) continue;
          #pragma unroll
          for (int vp = 0; vp < 3; ++vp) {
            const int rj = j + vp - 2;
            if (rj < 0 || rj >= KW) continue;
            const float t = wp[up * 3 + vp] * rp[ri * KW + rj];
            sF += t;
            if (up < 2) sU += t;
            if (vp < 2) sV += t;
            if (up < 2 && vp < 2) sUV += t;
          }
        }
      }
    }
    const size_t base = (size_t)(b * CI + ci);
    ws[(base * OC + oc) * KPAD + k] = (__bf16)sF;
    const size_t eb = (base * KPAD + k) * OC + oc;
    ws[VARSZ * 1 + eb] = (__bf16)sU;
    ws[VARSZ * 2 + eb] = (__bf16)sV;
    ws[VARSZ * 3 + eb] = (__bf16)sUV;
  }
}

// ---------------------------------------------------------------------------
// B-fragment im2col gather. With kBase and sel compile-time constants (sel via
// a lane-half branch at the call site), every (i,j) folds and each element is
// a ds_load_u16 at [base + immediate] -- no address VALU, no spills.
// ---------------------------------------------------------------------------
__device__ __forceinline__ v16bf buildB(const __bf16* impS, int baseElem,
                                        const int kBase, const int sel) {
  v16bf b;
  #pragma unroll
  for (int e = 0; e < 16; ++e) {
    int k = kBase + sel + (e & 7) + ((e >> 3) << 4);
    int i = (k * 205) >> 11;           // k / 10 for k < 1024
    int j = k - i * 10;
    if (k >= KTAP) { i = 0; j = 0; }   // padded K: A is zero, any finite B ok
    b[e] = impS[baseElem + (17 - i) * 26 + (9 - j)];
  }
  return b;
}

// ---------------------------------------------------------------------------
// Main fused implicit-GEMM kernel: per block, a 32row x 16col x 64oc output
// tile. 8 waves; wave w owns rows [4w, 4w+4). bf16 WMMA 16x16x32, f32 acc.
// K2 slice staging uses the Tensor Data Mover (one tensor_load_to_lds per ci).
// ---------------------------------------------------------------------------
__global__ __launch_bounds__(256) void main_kernel(const float* __restrict__ imp,
                                                   const float* __restrict__ convb,
                                                   const __bf16* __restrict__ K2,
                                                   float* __restrict__ out) {
  __shared__ __align__(16) __bf16 k2S[OC * KPAD];   // 24.5 KB, A operand slice
  __shared__ __align__(16) __bf16 impS[49 * 26];    // 2.5 KB, imp halo tile (bf16)

  const int x0   = blockIdx.x * 16;
  const int y0   = blockIdx.y * 32;
  const int b    = blockIdx.z;
  const int tid  = threadIdx.x;
  const int wave = tid >> 5;
  const int lane = tid & 31;
  const int n    = lane & 15;
  const int sel  = (lane >> 4) << 3;        // 0 (lanes 0-15) or 8 (lanes 16-31)
  const int aOffElem   = n * KPAD + sel;    // lane's A-row base (K-half included)
  const int bBaseElem0 = (wave << 2) * 26 + n;  // B gather base, t = 0

  v8f acc[4][4];
  #pragma unroll
  for (int t = 0; t < 4; ++t)
    #pragma unroll
    for (int o = 0; o < 4; ++o)
      acc[t][o] = (v8f){0.f, 0.f, 0.f, 0.f, 0.f, 0.f, 0.f, 0.f};

  for (int ci = 0; ci < CI; ++ci) {
    __syncthreads();
    // ---- Stage K2 slice [64][192] bf16 (24576 B, contiguous) via the TDM.
    // D# (ISA 8.3/8.4): 1-D tile of 6144 dwords; data_size=4B; type=2.
    if (wave == 0) {
      const unsigned ldsAddr = (unsigned)(size_t)(&k2S[0]);   // low 32b = LDS offset
      const unsigned long long ga =
          (unsigned long long)(const void*)(K2 + (size_t)(b * CI + ci) * OC * KPAD);
      const v4u g0 = { 1u,                                   // count=1, user mode
                       ldsAddr,                              // lds_addr
                       (unsigned)ga,                         // global_addr[31:0]
                       (unsigned)(ga >> 32) | 0x80000000u }; // ga[56:32] | type=2<<30
      const v8i g1 = { 0x20000,                // data_size=2 (4B), wg_mask=0
                       0x18000000,             // tensor_dim0 = 6144 (low16 << 16)
                       0x00010000,             // tensor_dim1 = 1
                       0x18000000,             // tile_dim0 = 6144 (<< 16)
                       0,                      // tile_dim1 = 0 (unused), tile_dim2 = 0
                       6144,                   // tensor_dim0_stride
                       0, 0 };
      const v4i g2 = { 0, 0, 0, 0 };
      const v4i g3 = { 0, 0, 0, 0 };
#if defined(__clang_major__) && (__clang_major__ >= 23)
      const v8i g4 = { 0, 0, 0, 0, 0, 0, 0, 0 };
      __builtin_amdgcn_tensor_load_to_lds(g0, g1, g2, g3, g4, 0);
#else
      __builtin_amdgcn_tensor_load_to_lds(g0, g1, g2, g3, 0);
#endif
      __builtin_amdgcn_s_wait_tensorcnt(0);
    }
    // ---- Stage impulse halo tile: rows [y0-16,y0+32], cols [x0-8,x0+16], bf16.
    {
      const float* ipBase = imp + (size_t)(b * CI + ci) * (HH * WW);
      for (int idx = tid; idx < 49 * 26; idx += 256) {
        const int r = idx / 26, c = idx % 26;
        const int gy = y0 - 16 + r, gx = x0 - 8 + c;
        float v = 0.f;
        if (c < 25 && (unsigned)gy < (unsigned)HH && (unsigned)gx < (unsigned)WW)
          v = ipBase[gy * WW + gx];
        impS[idx] = (__bf16)v;
      }
    }
    __syncthreads();

    const __bf16* aBase = k2S + aOffElem;
    #pragma unroll
    for (int chunk = 0; chunk < KPAD / 32; ++chunk) {
      const int kBase = chunk * 32;
      // A fragments: 2 x ds_load_b128 each, immediate offsets off one base.
      v16bf afr[4];
      #pragma unroll
      for (int ocT = 0; ocT < 4; ++ocT) {
        const __bf16* ap = aBase + ocT * 16 * KPAD + kBase;
        const v8bf lo = *(const v8bf*)ap;
        const v8bf hi = *(const v8bf*)(ap + 16);
        afr[ocT] = __builtin_shufflevector(lo, hi, 0, 1, 2, 3, 4, 5, 6, 7,
                                           8, 9, 10, 11, 12, 13, 14, 15);
      }
      // B fragments: lane-half branch makes sel literal -> immediate offsets.
      v16bf bfr[4];
      if (lane < 16) {
        #pragma unroll
        for (int t = 0; t < 4; ++t)
          bfr[t] = buildB(impS, bBaseElem0 + t * 26, kBase, 0);
      } else {
        #pragma unroll
        for (int t = 0; t < 4; ++t)
          bfr[t] = buildB(impS, bBaseElem0 + t * 26, kBase, 8);
      }
      // Converged: EXEC all ones for WMMA.
      #pragma unroll
      for (int t = 0; t < 4; ++t)
        #pragma unroll
        for (int ocT = 0; ocT < 4; ++ocT)
          acc[t][ocT] = __builtin_amdgcn_wmma_f32_16x16x32_bf16(
              false, afr[ocT], false, bfr[t], (short)0, acc[t][ocT], false, false);
    }
  }

  // Epilogue: C layout -> VGPR v: lanes0-15 M=v, lanes16-31 M=v+8; N=lane&15.
  const int px = x0 + n;
  #pragma unroll
  for (int t = 0; t < 4; ++t) {
    const int py = y0 + (wave << 2) + t;
    #pragma unroll
    for (int ocT = 0; ocT < 4; ++ocT) {
      #pragma unroll
      for (int v = 0; v < 8; ++v) {
        const int oc = ocT * 16 + sel + v;
        out[(((size_t)b * OC + oc) * HH + py) * WW + px] = acc[t][ocT][v] + convb[oc];
      }
    }
  }
}

// ---------------------------------------------------------------------------
// Edge fix-up: last row / last col differ from the merged formula because the
// reference zero-pads x beyond [0,H)x[0,W). Recompute those 767 pixels/sample
// exactly with the clipped-K2 variants. 64 threads = 64 oc, one pixel/block.
// ---------------------------------------------------------------------------
__global__ __launch_bounds__(64) void edge_kernel(const float* __restrict__ imp,
                                                  const float* __restrict__ convb,
                                                  const __bf16* __restrict__ ws,
                                                  float* __restrict__ out) {
  const int b = blockIdx.y;
  const int s = blockIdx.x;
  int py, px, var;
  if (s < WW) { py = HH - 1; px = s; var = (s == WW - 1) ? 3 : 1; }
  else        { py = s - WW; px = WW - 1; var = 2; }
  const int oc = threadIdx.x;
  float acc = convb[oc];
  for (int ci = 0; ci < CI; ++ci) {
    const __bf16* k2p = ws + (size_t)var * VARSZ + (size_t)(b * CI + ci) * KPAD * OC;
    const float*  ip  = imp + (size_t)(b * CI + ci) * HH * WW;
    for (int k = 0; k < KTAP; ++k) {
      const int i = (k * 205) >> 11;
      const int j = k - i * 10;
      const int gy = py + 1 - i, gx = px + 1 - j;
      float iv = 0.f;
      if ((unsigned)gy < (unsigned)HH && (unsigned)gx < (unsigned)WW)
        iv = ip[gy * WW + gx];
      acc += (float)k2p[k * OC + oc] * iv;
    }
  }
  out[(((size_t)b * OC + oc) * HH + py) * WW + px] = acc;
}

extern "C" void kernel_launch(void* const* d_in, const int* in_sizes, int n_in,
                              void* d_out, int out_size, void* d_ws, size_t ws_size,
                              hipStream_t stream) {
  (void)in_sizes; (void)n_in; (void)out_size; (void)ws_size;
  const float* resp = (const float*)d_in[0];   // [8,8,16,8]
  const float* imp  = (const float*)d_in[1];   // [8,16,256,512]
  const float* w    = (const float*)d_in[2];   // [64,128,3,3]
  const float* bias = (const float*)d_in[3];   // [64]
  float* out = (float*)d_out;                  // [8,64,256,512] f32
  __bf16* ws = (__bf16*)d_ws;                  // 4 K2 variants, ~12.6 MB

  k2_kernel<<<dim3(BB * CI), 256, 0, stream>>>(resp, w, ws);
  main_kernel<<<dim3(WW / 16, HH / 32, BB), 256, 0, stream>>>(imp, bias, ws, out);
  edge_kernel<<<dim3(WW + HH - 1, BB), 64, 0, stream>>>(imp, bias, ws, out);
}
